// Projector_16355235463833
// MI455X (gfx1250) — matrix operand order repaired and checked by reading, not compiled, not run on hardware
//
#include <hip/hip_runtime.h>
#include <hip/hip_bf16.h>
#include <cstdint>

// ---------------- CDNA5 WMMA / TDM types ----------------
typedef __attribute__((ext_vector_type(16))) _Float16 v16h;
typedef __attribute__((ext_vector_type(8)))  float    v8f;
typedef __attribute__((ext_vector_type(2)))  _Float16 h2v;
typedef __attribute__((ext_vector_type(4)))  unsigned int v4u;
typedef __attribute__((ext_vector_type(8)))  int      v8i;
typedef __attribute__((ext_vector_type(4)))  int      v4i;

// Problem constants
#define NB   16
#define LL   256
#define CC   512
#define HEAD 16
#define DD   32
#define HW   1024

#if defined(__has_builtin)
#if __has_builtin(__builtin_amdgcn_tensor_load_to_lds) && \
    __has_builtin(__builtin_amdgcn_s_wait_tensorcnt)
#define HAVE_TDM 1
#endif
#endif
#ifndef HAVE_TDM
#define HAVE_TDM 0
#endif

// A-operand (16x32 f16, MxK) half2 K-offset for vgpr-pair j (0..7), lane group g
__device__ __forceinline__ int a_k0(int j, int g) {
    return (j < 4) ? (2 * j + 8 * g) : (16 + 2 * (j - 4) + 8 * g);
}
// B-operand (32x16 f16, KxN) half2 K-offset for vgpr-pair j, lane group g
__device__ __forceinline__ int b_k0(int j, int g) { return (j < 4) ? (2 * j + 8 * g) : (16 + 2 * (j - 4) + 8 * g); }

#if HAVE_TDM
// Issue one TDM DMA: contiguous run of `elems` f16 from global -> LDS.
// Descriptor built per CDNA5 ISA D# layout (CDNA5 ISA).
__device__ __forceinline__ void tdm_load_f16(const _Float16* gptr,
                                             void* lds_ptr,
                                             unsigned int elems) {
  const unsigned long long ga = (unsigned long long)(uintptr_t)gptr;
  const unsigned int lds_off = (unsigned int)(uintptr_t)lds_ptr; // low 32b = LDS byte offset
  v4u g0;
  g0[0] = 1u;                                   // count=1, user descriptor, no gather
  g0[1] = lds_off;                              // lds_addr (bytes)
  g0[2] = (unsigned int)(ga & 0xffffffffu);     // global_addr[31:0]
  g0[3] = (unsigned int)((ga >> 32) & 0x01ffffffu) | 0x80000000u; // addr[56:32], type=2

  const unsigned int T0 = elems;  // tensor_dim0 (f16 units)
  const unsigned int S0 = elems;  // tensor_dim0_stride
  v8i g1;
  g1[0] = 0x00010000;                                   // data_size=1 (2 bytes)
  g1[1] = (int)((T0 & 0xffffu) << 16);                  // tensor_dim0[15:0] @ bits 63:48
  g1[2] = (int)(((T0 >> 16) & 0xffffu) | (1u << 16));   // tensor_dim0[31:16], tensor_dim1=1
  g1[3] = (int)((T0 & 0xffffu) << 16);                  // tensor_dim1[31:16]=0, tile_dim0
  g1[4] = 1;                                            // tile_dim1=1, tile_dim2=0
  g1[5] = (int)S0;                                      // tensor_dim0_stride[31:0]
  g1[6] = (int)(((S0 >> 32) & 0u) | ((S0 & 0xffffu) << 16)); // stride0 hi, stride1 lo
  g1[7] = (int)((S0 >> 16) & 0xffffffffu);              // tensor_dim1_stride hi bits

  v4i z4 = (v4i){0, 0, 0, 0};
#if __clang_major__ >= 23
  v8i z8 = (v8i){0, 0, 0, 0, 0, 0, 0, 0};
  __builtin_amdgcn_tensor_load_to_lds(g0, g1, z4, z4, z8, 0);
#else
  __builtin_amdgcn_tensor_load_to_lds(g0, g1, z4, z4, 0);
#endif
}
#endif // HAVE_TDM

// =====================================================================
// Projection GEMM: out[sp, c_out] = sum_c A[sp,c] * W[c_out,c] + bias
//   A = feature^T (SRC_TOKEN=false, staged via LDS transpose)  or
//   A = token     (SRC_TOKEN=true, direct global loads; (L,C) row-major)
// Block: 128 threads (4 waves). Tile: 64 spatial rows x 128 c_out cols.
// =====================================================================
template <int SP, bool SRC_TOKEN, bool STORE_T>
__global__ __launch_bounds__(128) void proj_kernel(
    const float* __restrict__ src, const float* __restrict__ w,
    const float* __restrict__ bias, _Float16* __restrict__ dst) {
  const int n = blockIdx.z;
  const int cout0 = blockIdx.y * 128;
  const int sp0 = blockIdx.x * 64;
  const int tid = threadIdx.x;
  const int lane = tid & 31;
  const int wv = tid >> 5;       // wave 0..3
  const int g = (lane >> 4) & 1; // lane group
  const int ln = lane & 15;

  __shared__ _Float16 bW[128 * 32]; // [c_out_local][c_local]  8KB
  __shared__ _Float16 aF[64 * 32];  // [p_local][c_local]      4KB

  v8f acc[8];
#pragma unroll
  for (int jt = 0; jt < 8; ++jt) acc[jt] = (v8f){0.f,0.f,0.f,0.f,0.f,0.f,0.f,0.f};

  for (int kc = 0; kc < 16; ++kc) {
    const int c0 = kc * 32;
    __syncthreads(); // protect previous iteration's LDS reads
    // ---- stage weight chunk: rows cout0..+128, cols c0..+32 ----
    {
      const float* wr = w + (size_t)(cout0 + tid) * CC + c0;
#pragma unroll
      for (int i = 0; i < 32; i += 4) {
        float4 f = *(const float4*)(wr + i);
        bW[tid * 32 + i + 0] = (_Float16)f.x;
        bW[tid * 32 + i + 1] = (_Float16)f.y;
        bW[tid * 32 + i + 2] = (_Float16)f.z;
        bW[tid * 32 + i + 3] = (_Float16)f.w;
      }
    }
    if (!SRC_TOKEN) {
      // stage feature chunk transposed: LDS[p][c], source (n, c, p)
      const int cl = tid >> 2;          // 0..31
      const int pb = (tid & 3) * 16;    // 0,16,32,48
      const float* fr = src + ((size_t)n * CC + (c0 + cl)) * SP + sp0 + pb;
#pragma unroll
      for (int i = 0; i < 16; i += 4) {
        float4 f = *(const float4*)(fr + i);
        aF[(pb + i + 0) * 32 + cl] = (_Float16)f.x;
        aF[(pb + i + 1) * 32 + cl] = (_Float16)f.y;
        aF[(pb + i + 2) * 32 + cl] = (_Float16)f.z;
        aF[(pb + i + 3) * 32 + cl] = (_Float16)f.w;
      }
    }
    __syncthreads();

    // ---- A operand: 16 spatial rows x 32 c ----
    v16h a;
    if (SRC_TOKEN) {
      const float* tr = src + ((size_t)n * SP + sp0 + wv * 16 + ln) * CC + c0;
#pragma unroll
      for (int j = 0; j < 8; ++j) {
        const int k0 = a_k0(j, g);
        float2 f = *(const float2*)(tr + k0);
        a[2 * j + 0] = (_Float16)f.x;
        a[2 * j + 1] = (_Float16)f.y;
      }
    } else {
      const _Float16* ar = &aF[(wv * 16 + ln) * 32];
#pragma unroll
      for (int j = 0; j < 8; ++j) {
        h2v h = *(const h2v*)(ar + a_k0(j, g));
        a[2 * j + 0] = h.x;
        a[2 * j + 1] = h.y;
      }
    }
    // ---- 8 c_out tiles ----
#pragma unroll
    for (int jt = 0; jt < 8; ++jt) {
      v16h b;
      const _Float16* br = &bW[(jt * 16 + ln) * 32];
#pragma unroll
      for (int j = 0; j < 8; ++j) {
        h2v h = *(const h2v*)(br + b_k0(j, g));
        b[2 * j + 0] = h.x;
        b[2 * j + 1] = h.y;
      }
      acc[jt] = __builtin_amdgcn_wmma_f32_16x16x32_f16(
          false, a, false, b, (short)0, acc[jt], false, false);
    }
  }

  // ---- bias + f16 store ----
#pragma unroll
  for (int jt = 0; jt < 8; ++jt) {
    const int co = cout0 + jt * 16 + ln;
    const float bv = bias[co];
    const int hh = co >> 5, dd = co & 31;
#pragma unroll
    for (int r = 0; r < 8; ++r) {
      const int sp = sp0 + wv * 16 + r + 8 * g;
      const float v = acc[jt][r] + bv;
      if (STORE_T)
        dst[(((size_t)n * HEAD + hh) * DD + dd) * SP + sp] = (_Float16)v;
      else
        dst[(((size_t)n * HEAD + hh) * SP + sp) * DD + dd] = (_Float16)v;
    }
  }
}

// =====================================================================
// Attention: per block (n, h, 128-query chunk), 256 threads = 8 waves.
// K/V staged to LDS via the Tensor Data Mover (TENSOR_LOAD_TO_LDS +
// s_wait_tensorcnt), 16 score WMMAs per wave, two-pass softmax through a
// per-wave LDS P-buffer, 16 WMMAs for P @ V^T, residual add epilogue.
// Dynamic LDS: K 16KB + V 16KB + 8x8KB P = 96KB.
// =====================================================================
__global__ __launch_bounds__(256) void attn_kernel(
    const float* __restrict__ feature, const _Float16* __restrict__ q_ws,
    const _Float16* __restrict__ k_ws, const _Float16* __restrict__ v_ws,
    float* __restrict__ out) {
  extern __shared__ __align__(16) _Float16 smem[];
  _Float16* kT = smem;             // [l][dd]  256*32
  _Float16* vT = smem + 8192;      // [dd][l]  32*256
  _Float16* pAll = smem + 16384;   // 8 waves * 16*256

  const int n = blockIdx.z, h = blockIdx.y;
  const int qb = blockIdx.x * 128;
  const int tid = threadIdx.x;
  const int lane = tid & 31;
  const int wv = tid >> 5; // 0..7
  const int g = (lane >> 4) & 1;
  const int ln = lane & 15;

  const _Float16* kSrc = k_ws + ((size_t)n * HEAD + h) * (LL * DD);
  const _Float16* vSrc = v_ws + ((size_t)n * HEAD + h) * (LL * DD);

#if HAVE_TDM
  // ---- TDM staging: wave0 DMAs K, wave1 DMAs V; each waits TENSORcnt ----
  if (wv == 0) {
    tdm_load_f16(kSrc, kT, LL * DD);
    __builtin_amdgcn_s_wait_tensorcnt(0);
  } else if (wv == 1) {
    tdm_load_f16(vSrc, vT, LL * DD);
    __builtin_amdgcn_s_wait_tensorcnt(0);
  }
#else
  // ---- fallback: coalesced 16B vector copies ----
  {
    const uint4* ks = (const uint4*)kSrc;
    const uint4* vs = (const uint4*)vSrc;
    uint4* kd = (uint4*)kT;
    uint4* vd = (uint4*)vT;
#pragma unroll
    for (int i = 0; i < 4; ++i) {
      kd[tid + i * 256] = ks[tid + i * 256];
      vd[tid + i * 256] = vs[tid + i * 256];
    }
  }
#endif
  __syncthreads();

  _Float16* pw = pAll + wv * (16 * 256);
  const int qrow = qb + wv * 16;

  // ---- Q operand (16 rows x 32 d) ----
  v16h aq;
  {
    const _Float16* qr = q_ws + (((size_t)n * HEAD + h) * HW + qrow + ln) * DD;
#pragma unroll
    for (int j = 0; j < 8; ++j) {
      h2v x = *(const h2v*)(qr + a_k0(j, g));
      aq[2 * j + 0] = x.x;
      aq[2 * j + 1] = x.y;
    }
  }

  const float inv = 0.17677669529663689f; // 1/sqrt(d)
  float m[8];
#pragma unroll
  for (int r = 0; r < 8; ++r) m[r] = -1e30f;

  // ---- pass 1: scores -> scaled f16 in LDS, track row max.
  //      Two independent WMMAs per step to hide the WMMA->VALU hazard. ----
#pragma unroll
  for (int jp = 0; jp < 8; ++jp) {
    const int colA = (2 * jp + 0) * 16 + ln;
    const int colB = (2 * jp + 1) * 16 + ln;
    v16h bkA, bkB;
    const _Float16* krA = &kT[colA * 32];
    const _Float16* krB = &kT[colB * 32];
#pragma unroll
    for (int j = 0; j < 8; ++j) {
      const int k0 = b_k0(j, g);
      h2v xA = *(const h2v*)(krA + k0);
      h2v xB = *(const h2v*)(krB + k0);
      bkA[2 * j + 0] = xA.x; bkA[2 * j + 1] = xA.y;
      bkB[2 * j + 0] = xB.x; bkB[2 * j + 1] = xB.y;
    }
    v8f z = (v8f){0.f,0.f,0.f,0.f,0.f,0.f,0.f,0.f};
    v8f sA = __builtin_amdgcn_wmma_f32_16x16x32_f16(
        false, aq, false, bkA, (short)0, z, false, false);
    v8f sB = __builtin_amdgcn_wmma_f32_16x16x32_f16(
        false, aq, false, bkB, (short)0, z, false, false);
#pragma unroll
    for (int r = 0; r < 8; ++r) {
      const float tA = sA[r] * inv;
      const float tB = sB[r] * inv;
      m[r] = fmaxf(m[r], fmaxf(tA, tB));
      pw[(r + 8 * g) * 256 + colA] = (_Float16)tA;
      pw[(r + 8 * g) * 256 + colB] = (_Float16)tB;
    }
  }
  // row-max across the 16 lanes sharing each row group
#pragma unroll
  for (int mask = 1; mask < 16; mask <<= 1)
#pragma unroll
    for (int r = 0; r < 8; ++r) m[r] = fmaxf(m[r], __shfl_xor(m[r], mask, 32));

  // ---- pass 2: exp + row sum, P back to LDS ----
  float sum[8];
#pragma unroll
  for (int r = 0; r < 8; ++r) sum[r] = 0.f;
#pragma unroll
  for (int jt = 0; jt < 16; ++jt) {
    const int col = jt * 16 + ln;
#pragma unroll
    for (int r = 0; r < 8; ++r) {
      const int idx = (r + 8 * g) * 256 + col;
      const float p = __expf((float)pw[idx] - m[r]);
      sum[r] += p;
      pw[idx] = (_Float16)p;
    }
  }
#pragma unroll
  for (int mask = 1; mask < 16; mask <<= 1)
#pragma unroll
    for (int r = 0; r < 8; ++r) sum[r] += __shfl_xor(sum[r], mask, 32);
  float rs[8];
#pragma unroll
  for (int r = 0; r < 8; ++r) rs[r] = 1.f / sum[r];

  __syncthreads();

  // ---- O = P(16x256) @ V^T(256x32): 8 K-chunks x 2 d-tiles ----
  v8f o0 = (v8f){0.f,0.f,0.f,0.f,0.f,0.f,0.f,0.f};
  v8f o1 = (v8f){0.f,0.f,0.f,0.f,0.f,0.f,0.f,0.f};
#pragma unroll
  for (int kc = 0; kc < 8; ++kc) {
    v16h ap;
    const _Float16* pr = pw + ln * 256 + kc * 32;
#pragma unroll
    for (int j = 0; j < 8; ++j) {
      h2v x = *(const h2v*)(pr + a_k0(j, g));
      ap[2 * j + 0] = x.x;
      ap[2 * j + 1] = x.y;
    }
    v16h bv0, bv1;
#pragma unroll
    for (int j = 0; j < 8; ++j) {
      const int k0 = b_k0(j, g);
      h2v x0 = *(const h2v*)(&vT[(0 + ln) * 256 + kc * 32 + k0]);
      h2v x1 = *(const h2v*)(&vT[(16 + ln) * 256 + kc * 32 + k0]);
      bv0[2 * j + 0] = x0.x; bv0[2 * j + 1] = x0.y;
      bv1[2 * j + 0] = x1.x; bv1[2 * j + 1] = x1.y;
    }
    o0 = __builtin_amdgcn_wmma_f32_16x16x32_f16(false, ap, false, bv0, (short)0, o0, false, false);
    o1 = __builtin_amdgcn_wmma_f32_16x16x32_f16(false, ap, false, bv1, (short)0, o1, false, false);
  }
#pragma unroll
  for (int r = 0; r < 8; ++r) { o0[r] *= rs[r]; o1[r] *= rs[r]; }

  // ---- epilogue: transpose O through LDS, coalesced residual add ----
  float* ob = (float*)pw; // reuse wave's P region (2KB of 8KB, 16B aligned)
  __syncthreads();
#pragma unroll
  for (int r = 0; r < 8; ++r) {
    ob[(0 + ln) * 16 + (r + 8 * g)] = o0[r];
    ob[(16 + ln) * 16 + (r + 8 * g)] = o1[r];
  }
  __syncthreads();

  const int dd = lane; // 0..31 channels of this head
  const size_t base = (((size_t)n * CC + h * DD + dd) * HW) + qrow;
  const float* fsrc = feature + base;
  float* dptr = out + base;
#pragma unroll
  for (int i = 0; i < 4; ++i) {
    float4 f = *(const float4*)(fsrc + i * 4);
    const float* ov = ob + dd * 16 + i * 4;
    float4 r;
    r.x = f.x + ov[0];
    r.y = f.y + ov[1];
    r.z = f.z + ov[2];
    r.w = f.w + ov[3];
    *(float4*)(dptr + i * 4) = r;
  }
}

// =====================================================================
extern "C" void kernel_launch(void* const* d_in, const int* in_sizes, int n_in,
                              void* d_out, int out_size, void* d_ws, size_t ws_size,
                              hipStream_t stream) {
  const float* feature = (const float*)d_in[0];
  const float* token   = (const float*)d_in[1];
  const float* wq = (const float*)d_in[2];
  const float* bq = (const float*)d_in[3];
  const float* wk = (const float*)d_in[4];
  const float* bk = (const float*)d_in[5];
  const float* wvp = (const float*)d_in[6];
  const float* bvp = (const float*)d_in[7];
  float* out = (float*)d_out;

  // workspace carve (f16): Q 16MB, K 4MB, V 4MB
  const size_t qElems = (size_t)NB * HEAD * HW * DD; // 8,388,608
  const size_t kElems = (size_t)NB * HEAD * LL * DD; // 2,097,152
  _Float16* q_ws = (_Float16*)d_ws;
  _Float16* k_ws = q_ws + qElems;
  _Float16* v_ws = k_ws + kElems;

  // Q = feature^T @ wq^T + bq -> (n,h,p,dd)
  proj_kernel<HW, false, false><<<dim3(HW / 64, CC / 128, NB), 128, 0, stream>>>(
      feature, wq, bq, q_ws);
  // K = token @ wk^T + bk -> (n,h,l,dd)
  proj_kernel<LL, true, false><<<dim3(LL / 64, CC / 128, NB), 128, 0, stream>>>(
      token, wk, bk, k_ws);
  // V = token @ wv^T + bv -> (n,h,dd,l)
  proj_kernel<LL, true, true><<<dim3(LL / 64, CC / 128, NB), 128, 0, stream>>>(
      token, wvp, bvp, v_ws);
  // attention + residual
  attn_kernel<<<dim3(HW / 128, HEAD, NB), 256, 96 * 1024, stream>>>(
      feature, q_ws, k_ws, v_ws, out);
}
